// Interaction_18227841204693
// MI455X (gfx1250) — matrix operand order repaired and checked by reading, not compiled, not run on hardware
//
#include <hip/hip_runtime.h>

typedef __attribute__((ext_vector_type(16))) _Float16 v16h;
typedef __attribute__((ext_vector_type(8)))  _Float16 v8h;
typedef __attribute__((ext_vector_type(8)))  float    v8f;

#define LDH 72            // f16 row pitch in LDS: 144B -> 36-bank stride, conflict-free frag loads
#define NTHREADS 256      // 8 wave32s per block

__device__ __forceinline__ float ssp(float x) {
  // softplus(x) - ln(2), numerically stable
  return fmaxf(x, 0.0f) + log1pf(__expf(-fabsf(x))) - 0.6931471805599453f;
}

// ---- WMMA fragment loads from LDS (layouts per CDNA5 ISA 7.12.2) -----------

// A: 16x32 f16 tile rows [row_base..row_base+15], K window [kb..kb+31]
__device__ __forceinline__ v16h load_A_frag(const _Float16* At, int row_base, int kb, int lane) {
  const int hi  = (lane >> 4) & 1;
  const int row = row_base + (lane & 15);
  const _Float16* p = At + row * LDH + kb + hi * 8;
  v8h lo = *(const v8h*)(p);        // K = kb + hi*8      .. +7
  v8h hh = *(const v8h*)(p + 16);   // K = kb + 16 + hi*8 .. +7
  v16h a;
#pragma unroll
  for (int e = 0; e < 8; ++e) { a[e] = lo[e]; a[e + 8] = hh[e]; }
  return a;
}

// B: 32x16 f16, built from row-major weight W[n][k]; cols [n_base..n_base+15]
__device__ __forceinline__ v16h load_B_frag(const _Float16* Wt, int n_base, int kb, int lane) {
  const int hi  = (lane >> 4) & 1;
  const int col = n_base + (lane & 15);
  const _Float16* p = Wt + col * LDH + kb + hi * 8;  // K = kb + hi*8 .. +7, kb + 16 + hi*8 .. +7
  v8h lo = *(const v8h*)(p);
  v8h hh = *(const v8h*)(p + 16);
  v16h b;
#pragma unroll
  for (int e = 0; e < 8; ++e) { b[e] = lo[e]; b[e + 8] = hh[e]; }
  return b;
}

// One wave computes a 16(M) x 64(N) strip: 4 C tiles x 2 K-steps = 8 WMMAs.
// acc initialized with per-column bias.
__device__ __forceinline__ void wave_gemm16x64(const _Float16* At, int row_base,
                                               const _Float16* Wt, const float* bias,
                                               int lane, v8f acc[4]) {
  v16h a0 = load_A_frag(At, row_base, 0,  lane);
  v16h a1 = load_A_frag(At, row_base, 32, lane);
#pragma unroll
  for (int t = 0; t < 4; ++t) {
    float bv = bias[t * 16 + (lane & 15)];
    v8f c;
#pragma unroll
    for (int r = 0; r < 8; ++r) c[r] = bv;
    v16h b0 = load_B_frag(Wt, t * 16, 0,  lane);
    v16h b1 = load_B_frag(Wt, t * 16, 32, lane);
    c = __builtin_amdgcn_wmma_f32_16x16x32_f16(false, a0, false, b0, (short)0, c, false, false);
    c = __builtin_amdgcn_wmma_f32_16x16x32_f16(false, a1, false, b1, (short)0, c, false, false);
    acc[t] = c;
  }
}

// C layout: VGPR r, lanes 0-15 -> (row_base+r, col), lanes 16-31 -> (row_base+8+r, col)
__device__ __forceinline__ void store_ssp_f16(_Float16* Tt, int row_base, int lane, const v8f acc[4]) {
  const int hi = (lane >> 4) & 1, cl = lane & 15;
#pragma unroll
  for (int t = 0; t < 4; ++t)
#pragma unroll
    for (int r = 0; r < 8; ++r)
      Tt[(row_base + r + hi * 8) * LDH + t * 16 + cl] = (_Float16)ssp(acc[t][r]);
}

// ---- cooperative global(f32) -> LDS(f16) staging ---------------------------

__device__ __forceinline__ void load_mat_f16(_Float16* dstLds, const float* src,
                                             int row0, int rmax, int tid) {
  for (int i = tid; i < 128 * 64; i += NTHREADS) {
    int r = i >> 6, c = i & 63;
    int gr = row0 + r;
    float v = (gr < rmax) ? src[(size_t)gr * 64 + c] : 0.0f;
    dstLds[r * LDH + c] = (_Float16)v;
  }
}
__device__ __forceinline__ void load_w_f16(_Float16* dstLds, const float* src, int tid) {
  for (int i = tid; i < 64 * 64; i += NTHREADS) {
    int r = i >> 6, c = i & 63;
    dstLds[r * LDH + c] = (_Float16)src[r * 64 + c];
  }
}
__device__ __forceinline__ void load_bias(float* dstLds, const float* src, int tid) {
  for (int i = tid; i < 64; i += NTHREADS) dstLds[i] = src[i];
}

// ---- kernels ---------------------------------------------------------------

__global__ void zero_kernel(float* __restrict__ p, int n) {
  int i = blockIdx.x * blockDim.x + threadIdx.x;
  if (i < n) p[i] = 0.0f;
}

// h = X @ Wn^T + bn   [V,64]
__global__ __launch_bounds__(NTHREADS)
void node_proj_kernel(const float* __restrict__ X, const float* __restrict__ Wn,
                      const float* __restrict__ bn, float* __restrict__ Hn, int V) {
  __shared__ __align__(16) _Float16 At[128 * LDH];
  __shared__ __align__(16) _Float16 Wt[64 * LDH];
  __shared__ float bs[64];
  const int tid = threadIdx.x;
  const int row0 = blockIdx.x * 128;
  load_mat_f16(At, X, row0, V, tid);
  load_w_f16(Wt, Wn, tid);
  load_bias(bs, bn, tid);
  __syncthreads();
  const int wave = tid >> 5, lane = tid & 31;
  v8f acc[4];
  wave_gemm16x64(At, wave * 16, Wt, bs, lane, acc);
  const int hi = (lane >> 4) & 1, cl = lane & 15;
#pragma unroll
  for (int t = 0; t < 4; ++t)
#pragma unroll
    for (int r = 0; r < 8; ++r) {
      int row = row0 + wave * 16 + r + hi * 8;
      if (row < V) Hn[(size_t)row * 64 + t * 16 + cl] = acc[t][r];
    }
}

// w = ssp(ssp(EF@W1^T+b1)@W2^T+b2); m = h[src]*w; agg[dst] += m
__global__ __launch_bounds__(NTHREADS)
void edge_mlp_scatter_kernel(const float* __restrict__ EF,
                             const int* __restrict__ src, const int* __restrict__ dst,
                             const float* __restrict__ W1, const float* __restrict__ b1,
                             const float* __restrict__ W2, const float* __restrict__ b2,
                             const float* __restrict__ Hn, float* __restrict__ Agg, int E) {
  __shared__ __align__(16) _Float16 At[128 * LDH];
  __shared__ __align__(16) _Float16 Tt[128 * LDH];
  __shared__ __align__(16) _Float16 W1t[64 * LDH];
  __shared__ __align__(16) _Float16 W2t[64 * LDH];
  __shared__ float b1s[64], b2s[64];
  __shared__ int srcs[128], dsts[128];
  const int tid = threadIdx.x;
  const int e0 = blockIdx.x * 128;

  // prefetch next tile's edge features toward L2 (global_prefetch_b8)
  if (e0 + 128 < E)
    __builtin_prefetch(EF + (size_t)(e0 + 128) * 64 + tid * 32, 0, 1);

  for (int i = tid; i < 128; i += NTHREADS) {
    int ge = e0 + i;
    srcs[i] = (ge < E) ? src[ge] : -1;
    dsts[i] = (ge < E) ? dst[ge] : 0;
  }
  load_mat_f16(At, EF, e0, E, tid);
  load_w_f16(W1t, W1, tid);
  load_w_f16(W2t, W2, tid);
  load_bias(b1s, b1, tid);
  load_bias(b2s, b2, tid);
  __syncthreads();

  const int wave = tid >> 5, lane = tid & 31;
  v8f acc[4];
  wave_gemm16x64(At, wave * 16, W1t, b1s, lane, acc);   // layer 1
  store_ssp_f16(Tt, wave * 16, lane, acc);
  __syncthreads();
  wave_gemm16x64(Tt, wave * 16, W2t, b2s, lane, acc);   // layer 2

  // fused: gather h[src] (L2-resident), multiply by ssp(acc), scatter-add
  const int hi = (lane >> 4) & 1, cl = lane & 15;
#pragma unroll
  for (int r = 0; r < 8; ++r) {
    int er = wave * 16 + r + hi * 8;
    int s = srcs[er], d = dsts[er];
    if (s < 0) continue;
#pragma unroll
    for (int t = 0; t < 4; ++t) {
      float w  = ssp(acc[t][r]);
      float hv = Hn[(size_t)s * 64 + t * 16 + cl];
      atomicAdd(&Agg[(size_t)d * 64 + t * 16 + cl], w * hv);
    }
  }
}

// out = ssp(Agg@Wc^T+bc) @ Wp^T + bp   (two fused WMMA GEMMs)
__global__ __launch_bounds__(NTHREADS)
void out_proj_kernel(const float* __restrict__ Agg,
                     const float* __restrict__ Wc, const float* __restrict__ bc,
                     const float* __restrict__ Wp, const float* __restrict__ bp,
                     float* __restrict__ Out, int V) {
  __shared__ __align__(16) _Float16 At[128 * LDH];
  __shared__ __align__(16) _Float16 Tt[128 * LDH];
  __shared__ __align__(16) _Float16 Wct[64 * LDH];
  __shared__ __align__(16) _Float16 Wpt[64 * LDH];
  __shared__ float bcs[64], bps[64];
  const int tid = threadIdx.x;
  const int row0 = blockIdx.x * 128;
  load_mat_f16(At, Agg, row0, V, tid);
  load_w_f16(Wct, Wc, tid);
  load_w_f16(Wpt, Wp, tid);
  load_bias(bcs, bc, tid);
  load_bias(bps, bp, tid);
  __syncthreads();
  const int wave = tid >> 5, lane = tid & 31;
  v8f acc[4];
  wave_gemm16x64(At, wave * 16, Wct, bcs, lane, acc);
  store_ssp_f16(Tt, wave * 16, lane, acc);
  __syncthreads();
  wave_gemm16x64(Tt, wave * 16, Wpt, bps, lane, acc);
  const int hi = (lane >> 4) & 1, cl = lane & 15;
#pragma unroll
  for (int t = 0; t < 4; ++t)
#pragma unroll
    for (int r = 0; r < 8; ++r) {
      int row = row0 + wave * 16 + r + hi * 8;
      if (row < V) Out[(size_t)row * 64 + t * 16 + cl] = acc[t][r];
    }
}

// ---- launcher --------------------------------------------------------------

extern "C" void kernel_launch(void* const* d_in, const int* in_sizes, int n_in,
                              void* d_out, int out_size, void* d_ws, size_t ws_size,
                              hipStream_t stream) {
  (void)n_in; (void)out_size; (void)ws_size;
  const float* node_feats = (const float*)d_in[0];
  const float* edge_feats = (const float*)d_in[1];
  const int*   src        = (const int*)d_in[2];
  const int*   dst        = (const int*)d_in[3];
  const float* Wn  = (const float*)d_in[4];
  const float* bn  = (const float*)d_in[5];
  const float* We1 = (const float*)d_in[6];
  const float* be1 = (const float*)d_in[7];
  const float* We2 = (const float*)d_in[8];
  const float* be2 = (const float*)d_in[9];
  const float* Wc  = (const float*)d_in[10];
  const float* bc  = (const float*)d_in[11];
  const float* Wp  = (const float*)d_in[12];
  const float* bp  = (const float*)d_in[13];

  const int V = in_sizes[0] / 64;
  const int E = in_sizes[2];

  float* Hn  = (float*)d_ws;                 // [V,64] projected node feats
  float* Agg = Hn + (size_t)V * 64;          // [V,64] scatter accumulator
  float* Out = (float*)d_out;

  zero_kernel<<<(V * 64 + 255) / 256, 256, 0, stream>>>(Agg, V * 64);
  node_proj_kernel<<<(V + 127) / 128, NTHREADS, 0, stream>>>(node_feats, Wn, bn, Hn, V);
  edge_mlp_scatter_kernel<<<(E + 127) / 128, NTHREADS, 0, stream>>>(
      edge_feats, src, dst, We1, be1, We2, be2, Hn, Agg, E);
  out_proj_kernel<<<(V + 127) / 128, NTHREADS, 0, stream>>>(Agg, Wc, bc, Wp, bp, Out, V);
}
